// LinearMultiheadAttention_5583457484859
// MI455X (gfx1250) — compile-verified, matrix operand order checked by reading, not run on hardware
//
#include <hip/hip_runtime.h>
#include <cstdint>

// ---------- types ----------
typedef __attribute__((ext_vector_type(16))) __bf16 v16bf;
typedef __attribute__((ext_vector_type(8)))  float  v8f;
typedef __attribute__((ext_vector_type(4)))  unsigned int v4u;

union FragU { v4u u[2]; v16bf v; };

static __device__ inline __bf16 f2bf(float f) {
  unsigned int u = __builtin_bit_cast(unsigned int, f);
  unsigned int r = (u + 0x7FFFu + ((u >> 16) & 1u)) >> 16;
  unsigned short s = (unsigned short)r;
  return __builtin_bit_cast(__bf16, s);
}
static __device__ inline float bf2f(__bf16 h) {
  unsigned short s = __builtin_bit_cast(unsigned short, h);
  unsigned int u = ((unsigned int)s) << 16;
  return __builtin_bit_cast(float, u);
}

// Load a 16x32 bf16 WMMA operand fragment (A-layout == B-layout on CDNA5):
// lane_lo = row (M or N), halves 0..7 hold K = hi*8 + 0..7,
// halves 8..15 hold K = 16 + hi*8 + 0..7  -> two b128 loads.
static __device__ inline v16bf load_frag(const __bf16* base, int ld, int rbase,
                                         int kbase, int lane_lo, int hi) {
  const __bf16* p = base + (size_t)(rbase + lane_lo) * ld + kbase + hi * 8;
  FragU f;
  f.u[0] = *(const v4u*)p;
  f.u[1] = *(const v4u*)(p + 16);
  return f.v;
}

static __device__ inline v8f wmma_bf16(v16bf a, v16bf b, v8f c) {
  return __builtin_amdgcn_wmma_f32_16x16x32_bf16(false, a, false, b,
                                                 (short)0, c, false, false);
}

// ---------- fp32 -> bf16 conversion ----------
__global__ void cvt_f32_bf16(const float* __restrict__ in,
                             __bf16* __restrict__ out, int n) {
  int i = blockIdx.x * blockDim.x + threadIdx.x;
  int stride = gridDim.x * blockDim.x;
  for (; i < n; i += stride) out[i] = f2bf(in[i]);
}

// ---------- GEMM: out = A (MxK) @ W^T (W is NxK row-major) + bias ----------
// mode 0: bf16 out; mode 1: exp epilogue, bf16 out; mode 2: fp32 out
__global__ __launch_bounds__(256) void gemm_xwT(const __bf16* __restrict__ A,
                                                const __bf16* __restrict__ W,
                                                const float* __restrict__ bias,
                                                __bf16* __restrict__ outh,
                                                float* __restrict__ outf,
                                                int M, int N, int K, int mode) {
  const int lane    = threadIdx.x & 31;
  const int wv      = threadIdx.x >> 5;
  const int lane_lo = lane & 15;
  const int hi      = lane >> 4;
  const int wm = wv >> 2, wn = wv & 3;          // 2 x 4 wave grid
  const int mbase = blockIdx.y * 64 + wm * 32;  // wave: 32 rows
  const int nbase = blockIdx.x * 256 + wn * 64; // wave: 64 cols

  v8f acc[2][4] = {};
  for (int kb = 0; kb < K; kb += 32) {
    v16bf a0 = load_frag(A, K, mbase,      kb, lane_lo, hi);
    v16bf a1 = load_frag(A, K, mbase + 16, kb, lane_lo, hi);
    v16bf b[4];
#pragma unroll
    for (int nt = 0; nt < 4; ++nt)
      b[nt] = load_frag(W, K, nbase + nt * 16, kb, lane_lo, hi);
#pragma unroll
    for (int nt = 0; nt < 4; ++nt) {
      acc[0][nt] = wmma_bf16(a0, b[nt], acc[0][nt]);
      acc[1][nt] = wmma_bf16(a1, b[nt], acc[1][nt]);
    }
  }
#pragma unroll
  for (int mt = 0; mt < 2; ++mt)
#pragma unroll
    for (int nt = 0; nt < 4; ++nt)
#pragma unroll
      for (int vi = 0; vi < 8; ++vi) {
        int row = mbase + mt * 16 + vi + hi * 8;
        int col = nbase + nt * 16 + lane_lo;
        float val = acc[mt][nt][vi] + bias[col];
        if (mode == 1) val = __expf(val);
        if (mode == 2) outf[(size_t)row * N + col] = val;
        else           outh[(size_t)row * N + col] = f2bf(val);
      }
}

// ==================== chunked linear attention, 3-pass parallel scan ==========
// B=2,S=2048,H=16,D=64; BH=32 sequences, 32 chunks of C=64 each.

// ---- Pass 1: per-(bh,chunk) local state: St[d2][d1] = sum_j V[j][d2]K[j][d1],
//              and local z[d] = sum_j K[j][d].  One WG per (chunk, bh), 4 waves.
__global__ __launch_bounds__(128) void attn_kv_local(const __bf16* __restrict__ k,
                                                     const __bf16* __restrict__ v,
                                                     float* __restrict__ kvbuf,
                                                     float* __restrict__ zbuf) {
  const int DIM = 1024, Sq = 2048;
  const int chunk = blockIdx.x, bh = blockIdx.y;
  const int bb = bh >> 4, h = bh & 15;
  const size_t headoff = (size_t)bb * Sq * DIM + (size_t)h * 64;
  const int s0 = chunk * 64;

  __shared__ __align__(16) __bf16 Kt[64 * 64];  // Kt[d][j] = K[j][d]
  __shared__ __align__(16) __bf16 Vt[64 * 64];  // Vt[d][j] = V[j][d]

  const int tid = threadIdx.x;
  const int lane = tid & 31, wv = tid >> 5;
  const int lane_lo = lane & 15, hi = lane >> 4;
  const int r0 = wv * 16;

  for (int idx = tid; idx < 4096; idx += 128) {
    int j = idx >> 6, d = idx & 63;
    Kt[d * 64 + j] = k[headoff + (size_t)(s0 + j) * DIM + d];
    Vt[d * 64 + j] = v[headoff + (size_t)(s0 + j) * DIM + d];
  }
  __syncthreads();

  v8f stacc[4] = {};
  v16bf va0 = load_frag(Vt, 64, r0, 0,  lane_lo, hi);
  v16bf va1 = load_frag(Vt, 64, r0, 32, lane_lo, hi);
#pragma unroll
  for (int d1t = 0; d1t < 4; ++d1t) {
    v16bf kb0 = load_frag(Kt, 64, d1t * 16, 0,  lane_lo, hi);
    v16bf kb1 = load_frag(Kt, 64, d1t * 16, 32, lane_lo, hi);
    stacc[d1t] = wmma_bf16(va0, kb0, stacc[d1t]);
    stacc[d1t] = wmma_bf16(va1, kb1, stacc[d1t]);
  }

  const size_t base = ((size_t)bh * 32 + chunk) * 4096;
#pragma unroll
  for (int d1t = 0; d1t < 4; ++d1t)
#pragma unroll
    for (int vi = 0; vi < 8; ++vi) {
      int d2 = r0 + vi + hi * 8;
      int d1 = d1t * 16 + lane_lo;
      kvbuf[base + d2 * 64 + d1] = stacc[d1t][vi];
    }
  if (tid < 64) {
    float s = 0.0f;
    for (int j = 0; j < 64; ++j) s += bf2f(Kt[tid * 64 + j]);
    zbuf[((size_t)bh * 32 + chunk) * 64 + tid] = s;
  }
}

// ---- Pass 2: in-place exclusive prefix over 32 chunks per bh. One WG per bh.
__global__ __launch_bounds__(128) void attn_scan(float* __restrict__ kvbuf,
                                                 float* __restrict__ zbuf) {
  const int bh = blockIdx.x;
  const int tid = threadIdx.x;
  float run[32];
#pragma unroll
  for (int i = 0; i < 32; ++i) run[i] = 0.0f;
  for (int c = 0; c < 32; ++c) {
    const size_t base = ((size_t)bh * 32 + c) * 4096;
#pragma unroll
    for (int i = 0; i < 32; ++i) {
      const size_t idx = base + tid + i * 128;
      float loc = kvbuf[idx];
      kvbuf[idx] = run[i];
      run[i] += loc;
    }
  }
  if (tid < 64) {
    float rz = 0.0f;
    for (int c = 0; c < 32; ++c) {
      const size_t zi = ((size_t)bh * 32 + c) * 64 + tid;
      float loc = zbuf[zi];
      zbuf[zi] = rz;
      rz += loc;
    }
  }
}

// ---- Pass 3: per-(bh,chunk) output:
//   A = QK^T (masked), num = A.V + Q.S_prefix, den = rowsum(A) + q.z_prefix
__global__ __launch_bounds__(128) void attn_out(const __bf16* __restrict__ q,
                                                const __bf16* __restrict__ k,
                                                const __bf16* __restrict__ v,
                                                const float* __restrict__ kvbuf,
                                                const float* __restrict__ zbuf,
                                                __bf16* __restrict__ o) {
  const int DIM = 1024, Sq = 2048;
  const int chunk = blockIdx.x, bh = blockIdx.y;
  const int bb = bh >> 4, h = bh & 15;
  const size_t headoff = (size_t)bb * Sq * DIM + (size_t)h * 64;
  const int s0 = chunk * 64;

  __shared__ __align__(16) __bf16 Qs[64 * 64];
  __shared__ __align__(16) __bf16 Ks[64 * 64];
  __shared__ __align__(16) __bf16 Vt[64 * 64];   // Vt[d][j] = V[j][d]
  __shared__ __align__(16) __bf16 As[64 * 64];   // masked QK^T
  __shared__ __align__(16) __bf16 Sts[64 * 64];  // Sts[d2][d1] = S_prefix[d1][d2]
  __shared__ float zsh[64];
  __shared__ float den[64];

  const int tid = threadIdx.x;
  const int lane = tid & 31, wv = tid >> 5;
  const int lane_lo = lane & 15, hi = lane >> 4;
  const int r0 = wv * 16;

  // stage Q, K row-major (b128 copies)
  for (int idx = tid; idx < 512; idx += 128) {
    int row = idx >> 3, seg = idx & 7;
    const v4u* sq = (const v4u*)(q + headoff + (size_t)(s0 + row) * DIM + seg * 8);
    const v4u* sk = (const v4u*)(k + headoff + (size_t)(s0 + row) * DIM + seg * 8);
    ((v4u*)Qs)[row * 8 + seg] = *sq;
    ((v4u*)Ks)[row * 8 + seg] = *sk;
  }
  // stage V^T and bf16 prefix state
  const size_t base = ((size_t)bh * 32 + chunk) * 4096;
  for (int idx = tid; idx < 4096; idx += 128) {
    int j = idx >> 6, d = idx & 63;
    Vt[d * 64 + j] = v[headoff + (size_t)(s0 + j) * DIM + d];
    Sts[idx] = f2bf(kvbuf[base + idx]);
  }
  if (tid < 64) {
    zsh[tid] = zbuf[((size_t)bh * 32 + chunk) * 64 + tid];
    den[tid] = 0.0f;
  }
  __syncthreads();

  // A = Q K^T strip, causal mask, write As
  v16bf qa0 = load_frag(Qs, 64, r0, 0,  lane_lo, hi);
  v16bf qa1 = load_frag(Qs, 64, r0, 32, lane_lo, hi);
  v8f accA[4] = {};
#pragma unroll
  for (int jt = 0; jt < 4; ++jt) {
    v16bf b0 = load_frag(Ks, 64, jt * 16, 0,  lane_lo, hi);
    v16bf b1 = load_frag(Ks, 64, jt * 16, 32, lane_lo, hi);
    accA[jt] = wmma_bf16(qa0, b0, accA[jt]);
    accA[jt] = wmma_bf16(qa1, b1, accA[jt]);
  }
#pragma unroll
  for (int jt = 0; jt < 4; ++jt)
#pragma unroll
    for (int vi = 0; vi < 8; ++vi) {
      int i = r0 + vi + hi * 8;
      int j = jt * 16 + lane_lo;
      float val = (j <= i) ? accA[jt][vi] : 0.0f;
      As[i * 64 + j] = f2bf(val);
    }
  __syncthreads();

  // den[i] = rowsum(A_masked[i,:]) + q_i . z_prefix   (2 threads per row)
  {
    int row = tid & 63, half = tid >> 6;
    float s = 0.0f;
    for (int j = half * 32; j < half * 32 + 32; ++j) {
      s += bf2f(As[row * 64 + j]);
      s += bf2f(Qs[row * 64 + j]) * zsh[j];
    }
    atomicAdd(&den[row], s);
  }

  // num = A_masked @ V + Q @ S_prefix
  v8f accN[4] = {};
  v16bf aa0 = load_frag(As, 64, r0, 0,  lane_lo, hi);
  v16bf aa1 = load_frag(As, 64, r0, 32, lane_lo, hi);
#pragma unroll
  for (int dt = 0; dt < 4; ++dt) {
    v16bf vb0 = load_frag(Vt, 64, dt * 16, 0,  lane_lo, hi);
    v16bf vb1 = load_frag(Vt, 64, dt * 16, 32, lane_lo, hi);
    accN[dt] = wmma_bf16(aa0, vb0, accN[dt]);
    accN[dt] = wmma_bf16(aa1, vb1, accN[dt]);
    v16bf sb0 = load_frag(Sts, 64, dt * 16, 0,  lane_lo, hi);
    v16bf sb1 = load_frag(Sts, 64, dt * 16, 32, lane_lo, hi);
    accN[dt] = wmma_bf16(qa0, sb0, accN[dt]);
    accN[dt] = wmma_bf16(qa1, sb1, accN[dt]);
  }
  __syncthreads();  // den atomics complete

#pragma unroll
  for (int dt = 0; dt < 4; ++dt)
#pragma unroll
    for (int vi = 0; vi < 8; ++vi) {
      int i  = r0 + vi + hi * 8;
      int d2 = dt * 16 + lane_lo;
      float val = accN[dt][vi] / den[i];
      o[headoff + (size_t)(s0 + i) * DIM + d2] = f2bf(val);
    }
}

// ---------- launch ----------
extern "C" void kernel_launch(void* const* d_in, const int* in_sizes, int n_in,
                              void* d_out, int out_size, void* d_ws, size_t ws_size,
                              hipStream_t stream) {
  (void)in_sizes; (void)n_in; (void)out_size; (void)ws_size;
  const float* x  = (const float*)d_in[0];
  const float* Wq = (const float*)d_in[1];
  const float* bq = (const float*)d_in[2];
  const float* Wk = (const float*)d_in[3];
  const float* bk = (const float*)d_in[4];
  const float* Wv = (const float*)d_in[5];
  const float* bv = (const float*)d_in[6];
  const float* Wo = (const float*)d_in[7];
  const float* bo = (const float*)d_in[8];
  float* out = (float*)d_out;

  const int M = 4096, N = 1024, K = 1024;  // M = B*S
  __bf16* p   = (__bf16*)d_ws;
  __bf16* xh  = p; p += (size_t)M * K;
  __bf16* wqh = p; p += (size_t)N * K;
  __bf16* wkh = p; p += (size_t)N * K;
  __bf16* wvh = p; p += (size_t)N * K;
  __bf16* woh = p; p += (size_t)N * K;
  __bf16* qb  = p; p += (size_t)M * N;
  __bf16* kb  = p; p += (size_t)M * N;
  __bf16* vb  = p; p += (size_t)M * N;
  __bf16* ab  = p; p += (size_t)M * N;
  float* kvbuf = (float*)p;                 // 32*32*4096 f32 = 16 MiB
  float* zbuf  = kvbuf + (size_t)32 * 32 * 4096;  // 32*32*64 f32

  cvt_f32_bf16<<<1024, 256, 0, stream>>>(x,  xh,  M * K);
  cvt_f32_bf16<<<512,  256, 0, stream>>>(Wq, wqh, N * K);
  cvt_f32_bf16<<<512,  256, 0, stream>>>(Wk, wkh, N * K);
  cvt_f32_bf16<<<512,  256, 0, stream>>>(Wv, wvh, N * K);
  cvt_f32_bf16<<<512,  256, 0, stream>>>(Wo, woh, N * K);

  dim3 g(N / 256, M / 64), blk(256);
  gemm_xwT<<<g, blk, 0, stream>>>(xh, wqh, bq, qb, nullptr, M, N, K, 1); // exp(q)
  gemm_xwT<<<g, blk, 0, stream>>>(xh, wkh, bk, kb, nullptr, M, N, K, 1); // exp(k)
  gemm_xwT<<<g, blk, 0, stream>>>(xh, wvh, bv, vb, nullptr, M, N, K, 0); // v

  attn_kv_local<<<dim3(32, 32), 128, 0, stream>>>(kb, vb, kvbuf, zbuf);
  attn_scan<<<32, 128, 0, stream>>>(kvbuf, zbuf);
  attn_out<<<dim3(32, 32), 128, 0, stream>>>(qb, kb, vb, kvbuf, zbuf, ab);

  gemm_xwT<<<g, blk, 0, stream>>>(ab, woh, bo, nullptr, out, M, N, K, 2); // fp32 out
}
